// MultiCodebookSoftVQ_23811298689883
// MI455X (gfx1250) — matrix operand order repaired and checked by reading, not compiled, not run on hardware
//
#include <hip/hip_runtime.h>
#include <hip/hip_bf16.h>
#include <math.h>

typedef __attribute__((ext_vector_type(16))) __bf16 v16bf;
typedef __attribute__((ext_vector_type(8)))  float  v8f;
typedef __attribute__((ext_vector_type(2)))  float  f32x2;

#define M_CB 8
#define K_CB 1024
#define D_CB 128
#define B_N  4
#define OUT_SAMPLE 4194304 // 4*1024*32*32
#define HALF_LOG_2PI 0.9189385332046727f
#define EPSV 1e-5f

// workspace layout (float units)
#define WS_KC     0          // [8192][4] packed per-code constants
#define WS_PART   32768      // 2048 KL partials
#define WS_FRAG   34816      // bf16 B-fragments: hi 2MB then lo 2MB (byte off 139264)
#define FRAG_LO_BYTES 2097152
#define N_WAVES   2048

// ---------------------------------------------------------------------------
// Kernel 0: per (m,k) constants packed as float4: {||mu||^2, 0.5/s^2,
// const logit term, normalized log prior}. One block per codebook m.
// ---------------------------------------------------------------------------
__global__ void vq_precompute(const float* __restrict__ mus,
                              const float* __restrict__ scales,
                              const float* __restrict__ logpy,
                              float* __restrict__ ws) {
  int m = blockIdx.x;
  int t = threadIdx.x;
  __shared__ float red[256];

  float mx = -3.0e38f;
  for (int k = t; k < K_CB; k += 256) mx = fmaxf(mx, logpy[m * K_CB + k]);
  red[t] = mx; __syncthreads();
  for (int s = 128; s > 0; s >>= 1) { if (t < s) red[t] = fmaxf(red[t], red[t + s]); __syncthreads(); }
  mx = red[0]; __syncthreads();
  float se = 0.f;
  for (int k = t; k < K_CB; k += 256) se += expf(logpy[m * K_CB + k] - mx);
  red[t] = se; __syncthreads();
  for (int s = 128; s > 0; s >>= 1) { if (t < s) red[t] += red[t + s]; __syncthreads(); }
  float lse = mx + logf(red[0]);

  for (int k = t; k < K_CB; k += 256) {
    int mk = m * K_CB + k;
    const float* mp = mus + (size_t)mk * D_CB;
    float sq = 0.f;
    for (int dd = 0; dd < D_CB; ++dd) { float v = mp[dd]; sq = fmaf(v, v, sq); }
    float s = fmaxf(scales[mk], EPSV);
    float lpy = logpy[mk];
    float4 c;
    c.x = sq;
    c.y = 0.5f / (s * s);
    c.z = -(float)D_CB * (logf(s) + HALF_LOG_2PI) + lpy;
    c.w = lpy - lse;
    *(float4*)(ws + WS_KC + (size_t)mk * 4) = c;
  }
}

// ---------------------------------------------------------------------------
// Kernel 0b: pre-convert mus fp32 -> bf16 hi/lo split, stored in per-lane
// WMMA B-fragment layout. One thread per (m, ktile, chunk, lane).
// ---------------------------------------------------------------------------
__global__ void vq_bfrag(const float* __restrict__ mus, float* __restrict__ ws) {
  int gid  = blockIdx.x * 256 + threadIdx.x;  // 0..65535 = m*8192 + kt*128 + c*32 + lane
  int lane = gid & 31;
  int c    = (gid >> 5) & 3;
  int kt   = (gid >> 7) & 63;
  int m    = gid >> 13;
  int half = lane >> 4;
  int row  = lane & 15;
  int kc   = kt * 16 + row;

  const float* bp = mus + (size_t)(m * K_CB + kc) * D_CB;
  v16bf hi, lo;
#pragma unroll
  for (int j = 0; j < 8; ++j) {
    // B 16-bit 32x16 layout: V0-3: K=(0..7)+8*half ; V4-7: K=16+(0..7)+8*half
    int kk = ((j < 4) ? (2 * j) : (16 + 2 * (j - 4))) + half * 8;
    int dd = c * 32 + kk;
    f32x2 p = *(const f32x2*)(bp + dd);
    __bf16 h0 = (__bf16)p.x, h1 = (__bf16)p.y;
    hi[2 * j]     = h0;
    hi[2 * j + 1] = h1;
    lo[2 * j]     = (__bf16)(p.x - (float)h0);
    lo[2 * j + 1] = (__bf16)(p.y - (float)h1);
  }
  __bf16* whi = (__bf16*)(ws + WS_FRAG);
  *(v16bf*)(whi + (size_t)gid * 16) = hi;
  *(v16bf*)(whi + 1048576 + (size_t)gid * 16) = lo;
}

// ---------------------------------------------------------------------------
// async global->LDS b128 (CDNA5), tracked with ASYNCcnt
// ---------------------------------------------------------------------------
__device__ __forceinline__ void async_b128(unsigned lds_addr, const void* sbase,
                                           unsigned voff) {
  asm volatile("global_load_async_to_lds_b128 %0, %1, %2"
               :: "v"(lds_addr), "v"(voff), "s"(sbase)
               : "memory");
}

// ---------------------------------------------------------------------------
// Kernel 1: main. Block = 8 waves, all on same codebook m (8 n-tiles).
// B fragments staged once per block via async-to-LDS double buffering.
// bf16x3 split GEMM via v_wmma_f32_16x16x32_bf16, fused streaming softmax.
// ---------------------------------------------------------------------------
__global__ __launch_bounds__(256) void vq_main(const float* __restrict__ x,
                                               const float* __restrict__ mus,
                                               const float* __restrict__ wsc,
                                               float* __restrict__ out,
                                               float* __restrict__ partials) {
  const int lane   = threadIdx.x & 31;
  const int wv     = threadIdx.x >> 5;
  const int m      = blockIdx.x & 7;
  const int ngroup = blockIdx.x >> 3;
  const int ntile  = ngroup * 8 + wv;
  const int n0     = ntile * 16;
  const int bi     = n0 >> 10;
  const int hw0    = n0 & 1023;
  const int half   = lane >> 4;
  const int row    = lane & 15;

  __shared__ alignas(32) __bf16 bstage[2][4096];  // [stage][hi 2048 | lo 2048]
  __shared__ float xsq_buf[8][16];
  __shared__ int   idx_buf[8][16];

  const char* fragbase = (const char*)(wsc + WS_FRAG);
  const unsigned t = threadIdx.x;

  auto issue = [&](int kt, int p) {
    unsigned so  = (unsigned)((m * 64 + kt) * 4096) + t * 16u;
    unsigned lhi = (unsigned)(uintptr_t)&bstage[p][t * 8u];
    unsigned llo = (unsigned)(uintptr_t)&bstage[p][2048 + t * 8u];
    async_b128(lhi, fragbase, so);
    async_b128(llo, fragbase, so + FRAG_LO_BYTES);
  };
  issue(0, 0);  // prologue: stage 0 in flight while we load A

  // ---- A fragments: x tile 16 rows x 128 d, fp32 -> bf16 hi/lo split ----
  const float* xb = x + (size_t)bi * 1048576 + (size_t)m * 131072 + hw0 + row;
  v16bf ahi[4], alo[4];
  float xsqp = 0.f;
#pragma unroll
  for (int c = 0; c < 4; ++c) {
#pragma unroll
    for (int j = 0; j < 8; ++j) {
      // A 16-bit 16x32 layout: V0-3: K=(0..7)+8*half ; V4-7: K=16+(0..7)+8*half
      int kk = ((j < 4) ? (2 * j) : (16 + 2 * (j - 4))) + half * 8;
      int dd = c * 32 + kk;
      float f0 = xb[(size_t)dd * 1024];
      float f1 = xb[(size_t)(dd + 1) * 1024];
      xsqp = fmaf(f0, f0, xsqp);
      xsqp = fmaf(f1, f1, xsqp);
      __bf16 h0 = (__bf16)f0, h1 = (__bf16)f1;
      ahi[c][2 * j]     = h0;
      ahi[c][2 * j + 1] = h1;
      alo[c][2 * j]     = (__bf16)(f0 - (float)h0);
      alo[c][2 * j + 1] = (__bf16)(f1 - (float)h1);
    }
  }
  xsqp += __shfl_xor(xsqp, 16);
  if (lane < 16) xsq_buf[wv][lane] = xsqp;
  __syncthreads();
  float xsq_r[8];
#pragma unroll
  for (int r = 0; r < 8; ++r) xsq_r[r] = xsq_buf[wv][r + half * 8];

  float mmax[8], Zs[8], As[8], Bs[8], bl[8];
  int bk[8];
#pragma unroll
  for (int r = 0; r < 8; ++r) {
    mmax[r] = -3.0e38f; Zs[r] = 0.f; As[r] = 0.f; Bs[r] = 0.f;
    bl[r] = -3.0e38f; bk[r] = 0;
  }

  const int mkbase = m * K_CB;
  const float4* kcbase = (const float4*)(wsc + WS_KC);

  for (int kt = 0; kt < 64; ++kt) {
    const int p = kt & 1;
    if (kt < 63) {
      issue(kt + 1, 1 - p);
      asm volatile("s_wait_asynccnt 2" ::: "memory");
    } else {
      asm volatile("s_wait_asynccnt 0" ::: "memory");
    }
    __syncthreads();  // whole stage buffer p is resident

    const int kc = kt * 16 + row;  // this lane's code column
    v8f acc = {0.f, 0.f, 0.f, 0.f, 0.f, 0.f, 0.f, 0.f};
#pragma unroll
    for (int c = 0; c < 4; ++c) {
      v16bf bhi = *(const v16bf*)&bstage[p][(c * 32 + lane) * 16];
      v16bf blo = *(const v16bf*)&bstage[p][2048 + (c * 32 + lane) * 16];
      acc = __builtin_amdgcn_wmma_f32_16x16x32_bf16(false, ahi[c], false, bhi,
                                                    (short)0, acc, false, false);
      acc = __builtin_amdgcn_wmma_f32_16x16x32_bf16(false, alo[c], false, bhi,
                                                    (short)0, acc, false, false);
      acc = __builtin_amdgcn_wmma_f32_16x16x32_bf16(false, ahi[c], false, blo,
                                                    (short)0, acc, false, false);
    }
    const float4 kcv = kcbase[mkbase + kc];
    const float musq = kcv.x, i2s = kcv.y, ct = kcv.z, lp = kcv.w;
#pragma unroll
    for (int r = 0; r < 8; ++r) {
      float dot = acc[r];
      float l = fmaf(2.f * dot - xsq_r[r] - musq, i2s, ct);
      bool better = (l > bl[r]);
      bk[r] = better ? kc : bk[r];
      bl[r] = better ? l : bl[r];
      float nm = fmaxf(mmax[r], l);
      float e1 = expf(mmax[r] - nm);
      float e2 = expf(l - nm);
      Zs[r] = Zs[r] * e1 + e2;
      As[r] = As[r] * e1 + l * e2;
      Bs[r] = Bs[r] * e1 + lp * e2;
      mmax[r] = nm;
    }
    __syncthreads();  // everyone done reading buffer p before it is re-issued
  }

  // merge the 16 lanes sharing each row
#pragma unroll
  for (int mask = 1; mask <= 8; mask <<= 1) {
#pragma unroll
    for (int r = 0; r < 8; ++r) {
      float om  = __shfl_xor(mmax[r], mask);
      float oZ  = __shfl_xor(Zs[r], mask);
      float oA  = __shfl_xor(As[r], mask);
      float oB  = __shfl_xor(Bs[r], mask);
      float obl = __shfl_xor(bl[r], mask);
      int   obk = __shfl_xor(bk[r], mask);
      float nm = fmaxf(mmax[r], om);
      float e1 = expf(mmax[r] - nm);
      float e2 = expf(om - nm);
      Zs[r] = Zs[r] * e1 + oZ * e2;
      As[r] = As[r] * e1 + oA * e2;
      Bs[r] = Bs[r] * e1 + oB * e2;
      mmax[r] = nm;
      bool take = (obl > bl[r]) || ((obl == bl[r]) && (obk < bk[r]));
      bl[r] = take ? obl : bl[r];
      bk[r] = take ? obk : bk[r];
    }
  }

  // KL partial: sum_k p*(l - L - lp) = A/Z - (mmax + logZ) - B/Z per row
  float klp = 0.f;
#pragma unroll
  for (int r = 0; r < 8; ++r) {
    float L = mmax[r] + logf(Zs[r]);
    klp += As[r] / Zs[r] - L - Bs[r] / Zs[r];
  }
  float klo = __shfl_xor(klp, 16);
  if (lane == 0) partials[blockIdx.x * 8 + wv] = klp + klo;

  if (row == 0) {
#pragma unroll
    for (int r = 0; r < 8; ++r) idx_buf[wv][half * 8 + r] = bk[r];
  }
  __syncthreads();

  // gather-scatter winning codes; for fixed dd, 16 lanes hit 16 consecutive hw
  float* ob = out + (size_t)bi * 1048576 + (size_t)m * 131072 + hw0 + row;
  int myidx = idx_buf[wv][row];
  const float* mv = mus + (size_t)m * (K_CB * D_CB) + (size_t)myidx * D_CB;
#pragma unroll 4
  for (int t2 = 0; t2 < 64; ++t2) {
    int dd = 2 * t2 + half;
    ob[(size_t)dd * 1024] = mv[dd];
  }
}

// ---------------------------------------------------------------------------
// Kernel 2: deterministic fixed-order reduction of KL partials.
// ---------------------------------------------------------------------------
__global__ void vq_finalize(const float* __restrict__ partials,
                            float* __restrict__ out_scalar) {
  __shared__ float red[256];
  int t = threadIdx.x;
  float s = 0.f;
  for (int i = t; i < N_WAVES; i += 256) s += partials[i];
  red[t] = s; __syncthreads();
  for (int k = 128; k > 0; k >>= 1) { if (t < k) red[t] += red[t + k]; __syncthreads(); }
  if (t == 0) {
    out_scalar[0] = red[0] / (float)B_N;
    out_scalar[1] = 0.0f;
  }
}

// ---------------------------------------------------------------------------
extern "C" void kernel_launch(void* const* d_in, const int* in_sizes, int n_in,
                              void* d_out, int out_size, void* d_ws, size_t ws_size,
                              hipStream_t stream) {
  const float* x      = (const float*)d_in[0];
  const float* mus    = (const float*)d_in[1];
  const float* scales = (const float*)d_in[2];
  const float* logpy  = (const float*)d_in[3];
  float* out = (float*)d_out;
  float* ws  = (float*)d_ws;

  vq_precompute<<<M_CB, 256, 0, stream>>>(mus, scales, logpy, ws);
  vq_bfrag<<<256, 256, 0, stream>>>(mus, ws);
  vq_main<<<256, 256, 0, stream>>>(x, mus, ws, out, ws + WS_PART);
  vq_finalize<<<1, 256, 0, stream>>>(ws + WS_PART, out + OUT_SAMPLE);
}
